// RelPositionMHA_3264175145120
// MI455X (gfx1250) — matrix operand order repaired and checked by reading, not compiled, not run on hardware
//
#include <hip/hip_runtime.h>
#include <hip/hip_bf16.h>
#include <math.h>

// MI455X / gfx1250, wave32. Transformer-XL relative-position MHA.
// B=4, T=2048, D=1024, H=8, DK=128, P=2T-1=4095.

typedef __attribute__((ext_vector_type(16))) __bf16 v16bf;
typedef __attribute__((ext_vector_type(8)))  __bf16 v8bf;
typedef __attribute__((ext_vector_type(8)))  float  v8f;
typedef __attribute__((ext_vector_type(4)))  unsigned int v4u;
typedef __attribute__((ext_vector_type(8)))  int v8i;
typedef __attribute__((ext_vector_type(4)))  int v4i;

#define Bn  4
#define Tn  2048
#define Dn  1024
#define Hn  8
#define DKn 128
#define Pn  (2 * Tn - 1)   // 4095

// Build a 16-element bf16 fragment from two contiguous 8-element (16B) runs.
__device__ __forceinline__ v16bf frag_from2(const __bf16* p0, const __bf16* p1) {
  v8bf a = *(const v8bf*)p0;
  v8bf b = *(const v8bf*)p1;
  v16bf r;
#pragma unroll
  for (int i = 0; i < 8; ++i) { r[i] = a[i]; r[8 + i] = b[i]; }
  return r;
}

__device__ __forceinline__ v8f wmma_bf16(v16bf a, v16bf b, v8f c) {
  return __builtin_amdgcn_wmma_f32_16x16x32_bf16(false, a, false, b, (short)0, c,
                                                 false, false);
}

// TDM: async 2D tile copy global -> LDS (row-major, padded pitch).
// tile: tile_d x tile_r elements of 2 bytes; global row stride = src_stride elems;
// LDS padded with 16B after each 256B row (pitch 136 elems for tile_d=128).
__device__ __forceinline__ void tdm_load_tile_2d(unsigned lds_off,
                                                 unsigned long long gaddr,
                                                 unsigned tensor_d0,
                                                 unsigned tensor_d1,
                                                 unsigned tile_d0,
                                                 unsigned tile_d1,
                                                 unsigned stride_d0) {
  v4u g0;
  g0[0] = 1u;                                   // count=1, user mode
  g0[1] = lds_off;                              // lds_addr [63:32]
  g0[2] = (unsigned)gaddr;                      // global_addr [95:64]
  g0[3] = ((unsigned)(gaddr >> 32) & 0x1FFFFFFu) | (2u << 30);  // [120:96], type=2
  v8i g1;
  // data_size=1 (2B) @ [17:16]; pad_enable @ [20]; pad_interval=5 (64 DW) @ [24:22];
  // pad_amount=3 (4 DW) @ [31:25]
  g1[0] = (int)((1u << 16) | (1u << 20) | (5u << 22) | (3u << 25));
  g1[1] = (int)((tensor_d0 & 0xFFFFu) << 16);           // tensor_dim0 [79:48] lo
  g1[2] = (int)(((tensor_d0 >> 16) & 0xFFFFu) |
                ((tensor_d1 & 0xFFFFu) << 16));          // dim0 hi | tensor_dim1 lo
  g1[3] = (int)(((tensor_d1 >> 16) & 0xFFFFu) |
                ((tile_d0 & 0xFFFFu) << 16));            // dim1 hi | tile_dim0
  g1[4] = (int)(tile_d1 & 0xFFFFu);                      // tile_dim1 | tile_dim2=0
  g1[5] = (int)stride_d0;                                // tensor_dim0_stride lo32
  g1[6] = 0;                                             // stride hi | dim1_stride lo
  g1[7] = 0;
  v4i z4 = {0, 0, 0, 0};
#if defined(__clang_major__) && (__clang_major__ >= 23)
  v8i z8 = {0, 0, 0, 0, 0, 0, 0, 0};
  __builtin_amdgcn_tensor_load_to_lds(g0, g1, z4, z4, z8, 0);
#else
  __builtin_amdgcn_tensor_load_to_lds(g0, g1, z4, z4, 0);
#endif
}

// ---------------- elementwise conversion ----------------
__global__ void cvt_f32_to_bf16(const float* __restrict__ s, __bf16* __restrict__ d,
                                int n) {
  int i = blockIdx.x * blockDim.x + threadIdx.x;
  if (i < n) d[i] = (__bf16)s[i];
}

// ---------------- pack kernels ----------------
__global__ void pack_q_kernel(const float* __restrict__ g, const float* __restrict__ u,
                              const float* __restrict__ v, __bf16* __restrict__ qu,
                              __bf16* __restrict__ qv) {
  int i = blockIdx.x * blockDim.x + threadIdx.x;
  if (i >= Bn * Tn * Dn) return;
  int c = i % Dn, tg = i / Dn;
  int h = c / DKn, d = c % DKn;
  int b = tg / Tn, t = tg % Tn;
  long o = (((long)(b * Hn + h) * Tn + t) * DKn + d);
  float q = g[i];
  qu[o] = (__bf16)(q + u[h * DKn + d]);
  qv[o] = (__bf16)(q + v[h * DKn + d]);
}

__global__ void pack_kv_kernel(const float* __restrict__ g, __bf16* __restrict__ o) {
  int i = blockIdx.x * blockDim.x + threadIdx.x;
  if (i >= Bn * Tn * Dn) return;
  int c = i % Dn, tg = i / Dn;
  int h = c / DKn, d = c % DKn;
  int b = tg / Tn, t = tg % Tn;
  o[(((long)(b * Hn + h) * Tn + t) * DKn + d)] = (__bf16)g[i];
}

__global__ void pack_p_kernel(const float* __restrict__ g, __bf16* __restrict__ o) {
  int i = blockIdx.x * blockDim.x + threadIdx.x;
  if (i >= Pn * Dn) return;
  int c = i % Dn, pr = i / Dn;
  int h = c / DKn, d = c % DKn;
  o[((long)h * Pn + pr) * DKn + d] = (__bf16)g[i];
}

// ---------------- bf16 WMMA GEMM: C[M,N] = A[M,K] * B[N,K]^T ----------------
__global__ __launch_bounds__(128) void gemm_nt_bf16(
    const __bf16* __restrict__ A, const __bf16* __restrict__ Bm,
    float* __restrict__ C, int M, int N, int K) {
  __shared__ __bf16 Bt[32][80];  // transposed B tile [k][n], 16B-aligned pitch

  const int tid = threadIdx.x, lane = tid & 31, w = tid >> 5;
  const int half = lane >> 4, n16 = lane & 15;
  const int m0 = blockIdx.x * 64 + w * 16;
  const int n0 = blockIdx.y * 64;

  v8f acc[4];
#pragma unroll
  for (int s = 0; s < 4; ++s) acc[s] = v8f{};

  int arow = m0 + n16;
  if (arow >= M) arow = M - 1;
  const __bf16* Arow = A + (long)arow * K;

  for (int kc = 0; kc < K; kc += 32) {
#pragma unroll
    for (int r = 0; r < 2; ++r) {
      int u = tid + r * 128;          // 256 vec8 units
      int n = u >> 2, seg = u & 3;
      v8bf bv = *(const v8bf*)(Bm + (long)(n0 + n) * K + kc + seg * 8);
#pragma unroll
      for (int i = 0; i < 8; ++i) Bt[seg * 8 + i][n] = bv[i];
    }
    __syncthreads();

    v16bf af = frag_from2(Arow + kc + 8 * half, Arow + kc + 16 + 8 * half);
#pragma unroll
    for (int s = 0; s < 4; ++s) {
      v16bf bf_;
      for (int i = 0; i < 8; ++i) { bf_[i] = Bt[8 * half + i][16 * s + n16]; bf_[8 + i] = Bt[16 + 8 * half + i][16 * s + n16]; }
      acc[s] = wmma_bf16(af, bf_, acc[s]);
    }
    __syncthreads();
  }

  float* Cbase = C + n0 + n16;
#pragma unroll
  for (int r = 0; r < 8; ++r) {
    int row = m0 + r + 8 * half;
    if (row < M) {
      float* Cr = Cbase + (long)row * N;
#pragma unroll
      for (int s = 0; s < 4; ++s) Cr[16 * s] = acc[s][r];
    }
  }
}

// ---------------- relative-position flash attention ----------------
// grid: (T/64, B*H); block: 128 threads (4 waves). Wave w: q-tile [qb+16w, +16).
// scores[q,k] = ((q+u).k + (q+v).p[T-1+k-q]) / sqrt(DK); softmax; O = P.V
__global__ __launch_bounds__(128) void relattn_kernel(
    const __bf16* __restrict__ QU, const __bf16* __restrict__ QV,
    const __bf16* __restrict__ Kh, const __bf16* __restrict__ Vh,
    const __bf16* __restrict__ Ph, float* __restrict__ Out) {
  __shared__ __bf16 Kt[128][40];     // K tile transposed [d][key]     10240 B
  __shared__ __bf16 Vt[32][136];     // V tile row-major (TDM-filled)   8704 B
  __shared__ __bf16 Pt[128][104];    // p window transposed [d][j]     26624 B
  __shared__ __bf16 Pr[4][16][40];   // per-wave prob tile [m][key]     5120 B
  __shared__ float  Dr[4][16][52];   // per-wave bd diag bank [m][jj]  13312 B

  const int tid = threadIdx.x, lane = tid & 31, w = tid >> 5;
  const int half = lane >> 4, n16 = lane & 15;
  const int bh = blockIdx.y;
  const int b = bh / Hn, h = bh % Hn;
  const int qb = blockIdx.x * 64;
  const int q0 = qb + w * 16;
  const float inv_sdk = 0.088388347648318447f;  // 1/sqrt(128)

  const __bf16* qurow = QU + ((long)bh * Tn + q0 + n16) * DKn;
  const __bf16* qvrow = QV + ((long)bh * Tn + q0 + n16) * DKn;
  v16bf qu[4], qv[4];
#pragma unroll
  for (int dc = 0; dc < 4; ++dc) {
    qu[dc] = frag_from2(qurow + dc * 32 + 8 * half, qurow + dc * 32 + 16 + 8 * half);
    qv[dc] = frag_from2(qvrow + dc * 32 + 8 * half, qvrow + dc * 32 + 16 + 8 * half);
  }

  v8f O[8];
#pragma unroll
  for (int i = 0; i < 8; ++i) O[i] = v8f{};
  float Mx[8], L[8];
#pragma unroll
  for (int r = 0; r < 8; ++r) { Mx[r] = -INFINITY; L[r] = 0.f; }

  const __bf16* Kbase = Kh + (long)bh * Tn * DKn;
  const __bf16* Vbase = Vh + (long)bh * Tn * DKn;
  const __bf16* Pbase = Ph + (long)h * Pn * DKn;
  const unsigned vt_lds = (unsigned)((size_t)&Vt[0][0] & 0xFFFFFFFFu);

  for (int k0 = 0; k0 < Tn; k0 += 32) {
    const int c0 = (Tn - 1) + k0 - qb;  // p-row center for wave 0

    // ---- TDM: async V tile (32 keys x 128 d, padded pitch 136) ----
    if (w == 0) {
      tdm_load_tile_2d(vt_lds,
                       (unsigned long long)(size_t)(Vbase + (long)k0 * DKn),
                       DKn, Tn, DKn, 32, DKn);
    }

    // ---- cooperative transposed staging of K and p-window ----
#pragma unroll
    for (int r = 0; r < 4; ++r) {       // 512 vec8 units: 32 keys x 16 d-segs
      int u = tid + r * 128;
      int key = u >> 4, dseg = u & 15;
      v8bf kv = *(const v8bf*)(Kbase + (long)(k0 + key) * DKn + dseg * 8);
#pragma unroll
      for (int i = 0; i < 8; ++i) Kt[dseg * 8 + i][key] = kv[i];
    }
#pragma unroll
    for (int r = 0; r < 12; ++r) {      // 1536 vec8 units: 96 p-rows x 16 d-segs
      int u = tid + r * 128;
      int j = u >> 4, dseg = u & 15;
      int prow = c0 - 63 + j;
      prow = prow < 0 ? 0 : (prow > Pn - 1 ? Pn - 1 : prow);
      v8bf pv = *(const v8bf*)(Pbase + (long)prow * DKn + dseg * 8);
#pragma unroll
      for (int i = 0; i < 8; ++i) Pt[dseg * 8 + i][j] = pv[i];
    }
    // prefetch next K chunk (global_prefetch_b8)
    if (k0 + 32 < Tn)
      __builtin_prefetch(Kbase + (long)(k0 + 32 + (tid >> 2)) * DKn + (tid & 3) * 32,
                         0, 1);
    if (w == 0) __builtin_amdgcn_s_wait_tensorcnt(0);  // V tile landed
    __syncthreads();

    // ---- bd tiles: Dr[w][m][jj] = Qv[m] . p[c_w - 15 + jj], jj in [0,48) ----
#pragma unroll
    for (int t3 = 0; t3 < 3; ++t3) {
      v8f acc = v8f{};
      const int jb = 48 - 16 * w + 16 * t3;
#pragma unroll
      for (int dc = 0; dc < 4; ++dc) {
        v16bf pf;
        for (int i = 0; i < 8; ++i) { pf[i] = Pt[dc * 32 + 8 * half + i][jb + n16]; pf[8 + i] = Pt[dc * 32 + 16 + 8 * half + i][jb + n16]; }
        acc = wmma_bf16(qv[dc], pf, acc);
      }
#pragma unroll
      for (int r = 0; r < 8; ++r) Dr[w][r + 8 * half][16 * t3 + n16] = acc[r];
    }

    // ---- content score tiles (two 16-key subtiles) ----
    v8f sa = v8f{}, sb = v8f{};
#pragma unroll
    for (int dc = 0; dc < 4; ++dc) {
      v16bf kfa;
      for (int i = 0; i < 8; ++i) { kfa[i] = Kt[dc * 32 + 8 * half + i][n16]; kfa[8 + i] = Kt[dc * 32 + 16 + 8 * half + i][n16]; }
      sa = wmma_bf16(qu[dc], kfa, sa);
      v16bf kfb;
      for (int i = 0; i < 8; ++i) { kfb[i] = Kt[dc * 32 + 8 * half + i][16 + n16]; kfb[8 + i] = Kt[dc * 32 + 16 + 8 * half + i][16 + n16]; }
      sb = wmma_bf16(qu[dc], kfb, sb);
    }

    // ---- online softmax over the 32-key chunk ----
    float alpha[8];
#pragma unroll
    for (int r = 0; r < 8; ++r) {
      const int row = r + 8 * half;
      float va = (sa[r] + Dr[w][row][15 + n16 - row]) * inv_sdk;
      float vb = (sb[r] + Dr[w][row][31 + n16 - row]) * inv_sdk;
      float cm = fmaxf(va, vb);
      cm = fmaxf(cm, __shfl_xor(cm, 1, 32));
      cm = fmaxf(cm, __shfl_xor(cm, 2, 32));
      cm = fmaxf(cm, __shfl_xor(cm, 4, 32));
      cm = fmaxf(cm, __shfl_xor(cm, 8, 32));
      float mn = fmaxf(Mx[r], cm);
      alpha[r] = __expf(Mx[r] - mn);
      Mx[r] = mn;
      float pA = __expf(va - mn);
      float pB = __expf(vb - mn);
      float rs = pA + pB;
      rs += __shfl_xor(rs, 1, 32);
      rs += __shfl_xor(rs, 2, 32);
      rs += __shfl_xor(rs, 4, 32);
      rs += __shfl_xor(rs, 8, 32);
      L[r] = L[r] * alpha[r] + rs;
      Pr[w][row][n16] = (__bf16)pA;
      Pr[w][row][16 + n16] = (__bf16)pB;
    }
#pragma unroll
    for (int nch = 0; nch < 8; ++nch)
#pragma unroll
      for (int r = 0; r < 8; ++r) O[nch][r] *= alpha[r];

    // ---- PV: probs (16x32) as A-fragment, V as B-fragments ----
    v16bf pfA = frag_from2(&Pr[w][n16][8 * half], &Pr[w][n16][16 + 8 * half]);
#pragma unroll
    for (int nch = 0; nch < 8; ++nch) {
      v16bf vf;
      for (int i = 0; i < 8; ++i) { vf[i] = Vt[8 * half + i][nch * 16 + n16]; vf[8 + i] = Vt[16 + 8 * half + i][nch * 16 + n16]; }
      O[nch] = wmma_bf16(pfA, vf, O[nch]);
    }
    __syncthreads();
  }

  // ---- normalize and store ----
  float* Obase = Out + ((long)(b * Tn + q0)) * Dn + h * DKn + n16;
#pragma unroll
  for (int r = 0; r < 8; ++r) {
    const int row = r + 8 * half;
    const float invl = 1.0f / L[r];
    float* Orow = Obase + (long)row * Dn;
#pragma unroll
    for (int nch = 0; nch < 8; ++nch) Orow[nch * 16] = O[nch][r] * invl;
  }
}

// ---------------- host launcher ----------------
extern "C" void kernel_launch(void* const* d_in, const int* in_sizes, int n_in,
                              void* d_out, int out_size, void* d_ws, size_t ws_size,
                              hipStream_t stream) {
  (void)in_sizes; (void)n_in; (void)out_size; (void)ws_size;
  const float* x  = (const float*)d_in[0];
  const float* pe = (const float*)d_in[1];
  const float* Wq = (const float*)d_in[2];
  const float* Wk = (const float*)d_in[3];
  const float* Wv = (const float*)d_in[4];
  const float* Wo = (const float*)d_in[5];
  const float* Wp = (const float*)d_in[6];
  const float* bu = (const float*)d_in[7];
  const float* bv = (const float*)d_in[8];
  float* out = (float*)d_out;

  char* ws = (char*)d_ws;
  size_t off = 0;
  auto carve = [&](size_t bytes) -> char* {
    char* p = ws + off;
    off += (bytes + 255) & ~(size_t)255;
    return p;
  };
  __bf16* xb   = (__bf16*)carve((size_t)Bn * Tn * Dn * 2);
  __bf16* pb   = (__bf16*)carve((size_t)Pn * Dn * 2);
  __bf16* Wqb  = (__bf16*)carve((size_t)Dn * Dn * 2);
  __bf16* Wkb  = (__bf16*)carve((size_t)Dn * Dn * 2);
  __bf16* Wvb  = (__bf16*)carve((size_t)Dn * Dn * 2);
  __bf16* Wob  = (__bf16*)carve((size_t)Dn * Dn * 2);
  __bf16* Wpb  = (__bf16*)carve((size_t)Dn * Dn * 2);
  float*  g32  = (float*)carve((size_t)Bn * Tn * Dn * 4);  // reused scratch
  __bf16* qub  = (__bf16*)carve((size_t)Bn * Hn * Tn * DKn * 2);
  __bf16* qvb  = (__bf16*)carve((size_t)Bn * Hn * Tn * DKn * 2);
  __bf16* khb  = (__bf16*)carve((size_t)Bn * Hn * Tn * DKn * 2);
  __bf16* vhb  = (__bf16*)carve((size_t)Bn * Hn * Tn * DKn * 2);
  __bf16* phb  = (__bf16*)carve((size_t)Hn * Pn * DKn * 2);
  __bf16* attnb = (__bf16*)carve((size_t)Bn * Tn * Dn * 2);

  const int NBT = Bn * Tn;  // 8192
  auto cvt = [&](const float* s, __bf16* d, long n) {
    cvt_f32_to_bf16<<<(int)((n + 255) / 256), 256, 0, stream>>>(s, d, (int)n);
  };

  // 1) fp32 -> bf16
  cvt(x, xb, (long)Bn * Tn * Dn);
  cvt(pe, pb, (long)Pn * Dn);
  cvt(Wq, Wqb, (long)Dn * Dn);
  cvt(Wk, Wkb, (long)Dn * Dn);
  cvt(Wv, Wvb, (long)Dn * Dn);
  cvt(Wo, Wob, (long)Dn * Dn);
  cvt(Wp, Wpb, (long)Dn * Dn);

  // 2) projections (C = A * W^T), packed to head-major bf16
  dim3 gproj(NBT / 64, Dn / 64);
  int npack = (Bn * Tn * Dn + 255) / 256;

  gemm_nt_bf16<<<gproj, 128, 0, stream>>>(xb, Wqb, g32, NBT, Dn, Dn);
  pack_q_kernel<<<npack, 256, 0, stream>>>(g32, bu, bv, qub, qvb);

  gemm_nt_bf16<<<gproj, 128, 0, stream>>>(xb, Wkb, g32, NBT, Dn, Dn);
  pack_kv_kernel<<<npack, 256, 0, stream>>>(g32, khb);

  gemm_nt_bf16<<<gproj, 128, 0, stream>>>(xb, Wvb, g32, NBT, Dn, Dn);
  pack_kv_kernel<<<npack, 256, 0, stream>>>(g32, vhb);

  dim3 gp((Pn + 63) / 64, Dn / 64);
  gemm_nt_bf16<<<gp, 128, 0, stream>>>(pb, Wpb, g32, Pn, Dn, Dn);
  pack_p_kernel<<<(Pn * Dn + 255) / 256, 256, 0, stream>>>(g32, phb);

  // 3) fused rel-pos flash attention -> g32 as [B*T, D] f32
  relattn_kernel<<<dim3(Tn / 64, Bn * Hn), 128, 0, stream>>>(qub, qvb, khb, vhb,
                                                             phb, g32);

  // 4) output projection: out = attn @ Wo^T
  cvt(g32, attnb, (long)Bn * Tn * Dn);
  gemm_nt_bf16<<<gproj, 128, 0, stream>>>(attnb, Wob, out, NBT, Dn, Dn);
}
